// Model_61624190763038
// MI455X (gfx1250) — compile-verified, matrix operand order checked by reading, not run on hardware
//
#include <hip/hip_runtime.h>

typedef __attribute__((ext_vector_type(16))) __bf16 v16bf;
typedef __attribute__((ext_vector_type(8)))  float  v8f;
typedef unsigned short u16;
typedef unsigned int   u32;

enum { D_IN = 96, D_MAIN = 512, D_BLOCK = 1024, BQ = 1024,
       N_CAND = 65536, N_CLASSES = 10, TQ = 64, CHUNK = 1024, CK = N_CAND / CHUNK,
       QT = BQ / TQ };

#define WMMA_BF16(a, b, c) \
  __builtin_amdgcn_wmma_f32_16x16x32_bf16(false, (a), false, (b), (short)0, (c), false, false)

__device__ __forceinline__ u16 f2bf(float f) {
  u32 u = __float_as_uint(f);
  u += 0x7FFFu + ((u >> 16) & 1u);          // round-to-nearest-even
  return (u16)(u >> 16);
}

union FragU { v16bf v; uint4 q[2]; };

// Load a 16x32 bf16 A/B fragment: per ISA layout, lane (hi = lane>>4) holds
// k = hi*8 .. hi*8+7 and k = 16+hi*8 .. 16+hi*8+7 of its row -> two b128 loads.
__device__ __forceinline__ v16bf load_frag(const u16* rowp, int hi) {
  FragU f;
  f.q[0] = *(const uint4*)(rowp + hi * 8);
  f.q[1] = *(const uint4*)(rowp + 16 + hi * 8);
  return f.v;
}

// ---------------- cast kernels ----------------
__global__ void cast_bf16_kernel(const float* __restrict__ in, u16* __restrict__ out, long n) {
  long i = (long)blockIdx.x * blockDim.x + threadIdx.x;
  long stride = (long)gridDim.x * blockDim.x;
  for (; i < n; i += stride) out[i] = f2bf(in[i]);
}

// in [K][N] f32 -> out [N][K] bf16  (weights pre-transposed for contiguous B-frags)
__global__ void cast_bf16_t_kernel(const float* __restrict__ in, u16* __restrict__ out,
                                   int K, int N) {
  long i = (long)blockIdx.x * blockDim.x + threadIdx.x;
  if (i >= (long)K * N) return;
  int n = (int)(i / K), k = (int)(i % K);
  out[i] = f2bf(in[(long)k * N + n]);
}

// ---------------- generic bf16 WMMA GEMM ----------------
// C = act(A[M x K] @ BT^T + bias), BT stored [N][K].
// 256 threads = 8 waves; block tile 64(M) x 64(N); wave tile 16 x 32.
__global__ void gemm_bf16_kernel(const u16* __restrict__ A, int lda,
                                 const u16* __restrict__ BT, int ldb,
                                 const float* __restrict__ bias,
                                 float* __restrict__ outF, int ldf,
                                 u16* __restrict__ outB, int ldob, float oscale, int relu,
                                 u16* __restrict__ outBT, int mtot,
                                 int K) {
  const int w = threadIdx.x >> 5, lane = threadIdx.x & 31;
  const int ll = lane & 15, hi = lane >> 4;
  const int m0 = blockIdx.y * 64 + (w & 3) * 16;
  const int n0 = blockIdx.x * 64 + (w >> 2) * 32;

  v8f acc0 = {}, acc1 = {};
  const u16* arow = A + (long)(m0 + ll) * lda;
  const u16* brow0 = BT + (long)(n0 + ll) * ldb;
  const u16* brow1 = BT + (long)(n0 + 16 + ll) * ldb;
  for (int k0 = 0; k0 < K; k0 += 32) {
    v16bf a  = load_frag(arow + k0, hi);
    v16bf b0 = load_frag(brow0 + k0, hi);
    v16bf b1 = load_frag(brow1 + k0, hi);
    acc0 = WMMA_BF16(a, b0, acc0);
    acc1 = WMMA_BF16(a, b1, acc1);
  }
#pragma unroll
  for (int t = 0; t < 2; t++) {
    v8f acc = t ? acc1 : acc0;
    const int col = n0 + t * 16 + ll;
    const float bv = bias ? bias[col] : 0.0f;
    float v[8];
#pragma unroll
    for (int r = 0; r < 8; r++) {
      float x = acc[r] + bv;
      if (relu) x = fmaxf(x, 0.0f);
      v[r] = x;
    }
#pragma unroll
    for (int r = 0; r < 8; r++) {
      const int row = m0 + r + hi * 8;
      if (outF) outF[(long)row * ldf + col] = v[r];
      if (outB) outB[(long)row * ldob + col] = f2bf(v[r] * oscale);
    }
    if (outBT) {  // D-layout gives each lane 8 consecutive rows of one column
      u16 p[8];
#pragma unroll
      for (int r = 0; r < 8; r++) p[r] = f2bf(v[r] * oscale);
      uint4 q; __builtin_memcpy(&q, p, 16);
      *(uint4*)(outBT + (long)col * mtot + m0 + hi * 8) = q;
    }
  }
}

// ---------------- fused split-K flash softmax-attention ----------------
// grid = (CK, QT); 256 threads. Qbf pre-scaled by D_MAIN^-0.5/TEMP.
// Per WG: TQ=64 query rows x CHUNK=1024 candidates.
__global__ void attn_kernel(const u16* __restrict__ Qbf,    // [BQ][D_MAIN]
                            const u16* __restrict__ Kbf,    // [N_CAND][D_MAIN]
                            const u16* __restrict__ VbfT,   // [D_MAIN][N_CAND]
                            float* __restrict__ Opart,      // [QT*CK][TQ][D_MAIN]
                            float* __restrict__ pm, float* __restrict__ pl) {
  extern __shared__ char smem[];
  u16*      Qs = (u16*)smem;                                   // 64 KB
  _Float16* Ss = (_Float16*)(smem + TQ * D_MAIN * 2);          // 128 KB (reused as bf16 P)
  u16*      Ps = (u16*)Ss;
  float*    sm = (float*)(smem + TQ * D_MAIN * 2 + TQ * CHUNK * 2);
  float*    sl = sm + TQ;

  const int ck = blockIdx.x, qt = blockIdx.y;
  const long cbase = (long)ck * CHUNK;
  const int tid = threadIdx.x;
  const int w = tid >> 5, lane = tid & 31, ll = lane & 15, hi = lane >> 4;
  const int m0 = (w & 3) * 16;          // 4 M-strips of 16 rows

  {  // stage Q tile to LDS
    const uint4* src = (const uint4*)(Qbf + (long)qt * TQ * D_MAIN);
    uint4* dst = (uint4*)Qs;
    for (int i = tid; i < TQ * D_MAIN / 8; i += 256) dst[i] = src[i];
  }
  __syncthreads();

  // Phase A: S = Q K^T over this chunk; 8 waves tile 4(M) x 2(N) per 32-cand step
  const int nw = (w >> 2) * 16;
  for (int s = 0; s < CHUNK; s += 32) {
    const int n0 = s + nw;
    v8f c = {};
    const u16* qrow = Qs + (m0 + ll) * D_MAIN;
    const u16* krow = Kbf + (cbase + n0 + ll) * D_MAIN;
    __builtin_prefetch(Kbf + (cbase + n0 + 32 + ll) * D_MAIN, 0, 0);
    for (int k0 = 0; k0 < D_MAIN; k0 += 32) {
      v16bf a = load_frag(qrow + k0, hi);
      v16bf b = load_frag(krow + k0, hi);
      c = WMMA_BF16(a, b, c);
    }
#pragma unroll
    for (int r = 0; r < 8; r++)
      Ss[(m0 + r + hi * 8) * CHUNK + n0 + ll] = (_Float16)c[r];
  }
  __syncthreads();

  // Phase B: per-row max + sumexp (4 lanes per row), then S -> P = exp(S-m) as bf16
  {
    const int row = tid >> 2, seg = tid & 3;
    const _Float16* sp = Ss + row * CHUNK + seg * 256;
    float mx = -1e30f;
    for (int j = 0; j < 256; j++) mx = fmaxf(mx, (float)sp[j]);
    for (int d = 1; d < 4; d <<= 1) mx = fmaxf(mx, __shfl_xor(mx, d));
    float sum = 0.0f;
    for (int j = 0; j < 256; j++) sum += __expf((float)sp[j] - mx);
    for (int d = 1; d < 4; d <<= 1) sum += __shfl_xor(sum, d);
    if (seg == 0) {
      sm[row] = mx; sl[row] = sum;
      const long pidx = ((long)qt * CK + ck) * TQ + row;
      pm[pidx] = mx; pl[pidx] = sum;
    }
  }
  __syncthreads();
  for (int e = tid; e < TQ * CHUNK; e += 256) {
    const int row = e >> 10;  // CHUNK == 1024
    Ps[e] = f2bf(__expf((float)Ss[e] - sm[row]));
  }
  __syncthreads();

  // Phase C: O = P V ; wave covers its 16-row strip x 256 cols, in two
  // 128-wide passes to keep accumulators at 8 v8f (64 VGPRs).
  float* op = Opart + ((long)qt * CK + ck) * TQ * D_MAIN;
#pragma unroll
  for (int pass = 0; pass < 2; pass++) {
    const int d0 = (w >> 2) * 256 + pass * 128;
    v8f o[8];
    const v8f zf = {};
#pragma unroll
    for (int t = 0; t < 8; t++) o[t] = zf;
    for (int k0 = 0; k0 < CHUNK; k0 += 32) {
      v16bf a = load_frag(Ps + (m0 + ll) * CHUNK + k0, hi);
#pragma unroll
      for (int t = 0; t < 8; t++) {
        v16bf b = load_frag(VbfT + (long)(d0 + t * 16 + ll) * N_CAND + cbase + k0, hi);
        o[t] = WMMA_BF16(a, b, o[t]);
      }
    }
#pragma unroll
    for (int t = 0; t < 8; t++)
#pragma unroll
      for (int r = 0; r < 8; r++)
        op[(m0 + r + hi * 8) * D_MAIN + d0 + t * 16 + ll] = o[t][r];
  }
}

// ---------------- combine partials (log-sum-exp merge) ----------------
__global__ void combine_kernel(const float* __restrict__ Opart,
                               const float* __restrict__ pm, const float* __restrict__ pl,
                               float* __restrict__ ctx) {
  const int b = blockIdx.x, qt = b / TQ, row = b % TQ;
  float M = -1e30f;
  for (int ck = 0; ck < CK; ck++) M = fmaxf(M, pm[((long)qt * CK + ck) * TQ + row]);
  float L = 0.0f;
  for (int ck = 0; ck < CK; ck++) {
    const long idx = ((long)qt * CK + ck) * TQ + row;
    L += pl[idx] * __expf(pm[idx] - M);
  }
  const float inv = 1.0f / L;
  for (int d = threadIdx.x; d < D_MAIN; d += blockDim.x) {
    float acc = 0.0f;
    for (int ck = 0; ck < CK; ck++) {
      const long idx = ((long)qt * CK + ck) * TQ + row;
      acc += Opart[idx * D_MAIN + d] * __expf(pm[idx] - M);
    }
    ctx[(long)b * D_MAIN + d] = acc * inv;
  }
}

// ---------------- head: out = (query + ctx) @ head_w + head_b ----------------
__global__ void head_kernel(const float* __restrict__ query, const float* __restrict__ ctx,
                            const float* __restrict__ hw, const float* __restrict__ hb,
                            float* __restrict__ out) {
  const int b = blockIdx.x, c = threadIdx.x;
  if (c >= N_CLASSES) return;
  float acc = hb[c];
  for (int d = 0; d < D_MAIN; d++)
    acc += (query[(long)b * D_MAIN + d] + ctx[(long)b * D_MAIN + d]) * hw[(long)d * N_CLASSES + c];
  out[(long)b * N_CLASSES + c] = acc;
}

extern "C" void kernel_launch(void* const* d_in, const int* in_sizes, int n_in,
                              void* d_out, int out_size, void* d_ws, size_t ws_size,
                              hipStream_t stream) {
  const float* x_num   = (const float*)d_in[0];
  const float* cand_k  = (const float*)d_in[1];
  const float* enc_w0  = (const float*)d_in[2];
  const float* enc_b0  = (const float*)d_in[3];
  const float* enc_w1  = (const float*)d_in[4];
  const float* enc_b1  = (const float*)d_in[5];
  const float* enc_w2  = (const float*)d_in[6];
  const float* enc_b2  = (const float*)d_in[7];
  const float* trans_w = (const float*)d_in[8];
  const float* trans_b = (const float*)d_in[9];
  const float* head_w  = (const float*)d_in[10];
  const float* head_b  = (const float*)d_in[11];
  float* out = (float*)d_out;

  char* ws = (char*)d_ws;
  size_t off = 0;
  auto alloc = [&](size_t bytes) -> char* {
    char* p = ws + off;
    off = (off + bytes + 255) & ~(size_t)255;
    return p;
  };
  u16* Kbf   = (u16*)alloc((size_t)N_CAND * D_MAIN * 2);
  u16* VbfT  = (u16*)alloc((size_t)D_MAIN * N_CAND * 2);
  u16* W0T   = (u16*)alloc((size_t)D_MAIN * D_IN * 2);
  u16* W1T   = (u16*)alloc((size_t)2 * D_BLOCK * D_MAIN * 2);
  u16* W2T   = (u16*)alloc((size_t)2 * D_MAIN * D_BLOCK * 2);
  u16* TWT   = (u16*)alloc((size_t)D_MAIN * D_MAIN * 2);
  u16* Xbf   = (u16*)alloc((size_t)BQ * D_IN * 2);
  u16* Q0bf  = (u16*)alloc((size_t)BQ * D_MAIN * 2);
  u16* Hbf   = (u16*)alloc((size_t)BQ * D_BLOCK * 2);
  u16* Q1bf  = (u16*)alloc((size_t)BQ * D_MAIN * 2);
  u16* QbfS  = (u16*)alloc((size_t)BQ * D_MAIN * 2);
  float* queryF = (float*)alloc((size_t)BQ * D_MAIN * 4);
  float* Opart  = (float*)alloc((size_t)QT * CK * TQ * D_MAIN * 4);
  float* pmB    = (float*)alloc((size_t)QT * CK * TQ * 4);
  float* plB    = (float*)alloc((size_t)QT * CK * TQ * 4);
  float* ctx    = (float*)alloc((size_t)BQ * D_MAIN * 4);

  const float att_scale = 0.22097086912079612f;  // 512^-0.5 / 0.2

  // casts
  cast_bf16_kernel<<<4096, 256, 0, stream>>>(cand_k, Kbf, (long)N_CAND * D_MAIN);
  cast_bf16_kernel<<<256, 256, 0, stream>>>(x_num, Xbf, (long)BQ * D_IN);
  cast_bf16_t_kernel<<<(D_IN * D_MAIN + 255) / 256, 256, 0, stream>>>(enc_w0, W0T, D_IN, D_MAIN);
  for (int i = 0; i < 2; i++) {
    cast_bf16_t_kernel<<<(D_MAIN * D_BLOCK + 255) / 256, 256, 0, stream>>>(
        enc_w1 + (size_t)i * D_MAIN * D_BLOCK, W1T + (size_t)i * D_BLOCK * D_MAIN, D_MAIN, D_BLOCK);
    cast_bf16_t_kernel<<<(D_BLOCK * D_MAIN + 255) / 256, 256, 0, stream>>>(
        enc_w2 + (size_t)i * D_BLOCK * D_MAIN, W2T + (size_t)i * D_MAIN * D_BLOCK, D_BLOCK, D_MAIN);
  }
  cast_bf16_t_kernel<<<(D_MAIN * D_MAIN + 255) / 256, 256, 0, stream>>>(trans_w, TWT, D_MAIN, D_MAIN);

  // encoder: x @ W0 + b0
  gemm_bf16_kernel<<<dim3(D_MAIN / 64, BQ / 64), 256, 0, stream>>>(
      Xbf, D_IN, W0T, D_IN, enc_b0, nullptr, 0, Q0bf, D_MAIN, 1.0f, 0, nullptr, 0, D_IN);
  const u16* qin = Q0bf;
  for (int i = 0; i < 2; i++) {
    gemm_bf16_kernel<<<dim3(D_BLOCK / 64, BQ / 64), 256, 0, stream>>>(
        qin, D_MAIN, W1T + (size_t)i * D_BLOCK * D_MAIN, D_MAIN, enc_b1 + (size_t)i * D_BLOCK,
        nullptr, 0, Hbf, D_BLOCK, 1.0f, 1, nullptr, 0, D_MAIN);
    const bool last = (i == 1);
    gemm_bf16_kernel<<<dim3(D_MAIN / 64, BQ / 64), 256, 0, stream>>>(
        Hbf, D_BLOCK, W2T + (size_t)i * D_MAIN * D_BLOCK, D_BLOCK, enc_b2 + (size_t)i * D_MAIN,
        last ? queryF : nullptr, D_MAIN, last ? QbfS : Q1bf, D_MAIN,
        last ? att_scale : 1.0f, 0, nullptr, 0, D_BLOCK);
    qin = Q1bf;
  }

  // context values V = cand_k @ trans_w + trans_b, stored transposed bf16 [D_MAIN][N_CAND]
  gemm_bf16_kernel<<<dim3(D_MAIN / 64, N_CAND / 64), 256, 0, stream>>>(
      Kbf, D_MAIN, TWT, D_MAIN, trans_b, nullptr, 0, nullptr, 0, 1.0f, 0, VbfT, N_CAND, D_MAIN);

  // fused flash softmax-attention, split over candidate chunks
  const size_t smem = (size_t)TQ * D_MAIN * 2 + (size_t)TQ * CHUNK * 2 + 2 * TQ * 4;
  attn_kernel<<<dim3(CK, QT), 256, smem, stream>>>(QbfS, Kbf, VbfT, Opart, pmB, plB);

  combine_kernel<<<BQ, 256, 0, stream>>>(Opart, pmB, plB, ctx);
  head_kernel<<<BQ, 32, 0, stream>>>(queryF, ctx, head_w, head_b, out);
}
